// recurrent_ref_1030792151362
// MI455X (gfx1250) — compile-verified, matrix operand order checked by reading, not run on hardware
//
#include <hip/hip_runtime.h>
#include <hip/hip_bf16.h>

#define B_      4
#define L_      2048
#define W_      2560
#define LRU_    2560
#define HEADS_  8
#define BW_     320
#define DCONV_  4
#define ML_     (B_ * L_)        // 8192 rows
#define PAD     40               // LDS row pitch (elements) to spread banks
#define CLEN    128              // scan chunk length
#define NCH     16               // L_/CLEN chunks

typedef __attribute__((ext_vector_type(16))) __bf16 v16bf;
typedef __attribute__((ext_vector_type(8)))  __bf16 v8bf;
typedef __attribute__((ext_vector_type(8)))  float  v8f;

// ---------------- CDNA5 async global->LDS (CDNA5 ISA) ----------------
__device__ __forceinline__ void async_ld16(unsigned ldsoff, const void* gaddr) {
  asm volatile("global_load_async_to_lds_b128 %0, %1, off"
               :: "v"(ldsoff), "v"(gaddr) : "memory");
}
#define WAIT_ASYNC(n) asm volatile("s_wait_asynccnt %0" :: "n"(n) : "memory")

__device__ __forceinline__ v8f vzero() {
  v8f v;
#pragma unroll
  for (int i = 0; i < 8; ++i) v[i] = 0.0f;
  return v;
}

// Load one wave32 WMMA bf16 fragment from an LDS row-slice.
// Per CDNA5 ISA 16-bit A 16x32 layout: lanes 0-15 hold K {0..7,16..23},
// lanes 16-31 hold K {8..15,24..31}; B is stored N-major so it loads the same.
__device__ __forceinline__ v16bf ldfrag(const __bf16* p) {
  v8bf lo = *(const v8bf*)(p);
  v8bf hi = *(const v8bf*)(p + 16);
  v16bf r;
#pragma unroll
  for (int i = 0; i < 8; ++i) { r[i] = lo[i]; r[8 + i] = hi[i]; }
  return r;
}

// ------------------------------- bf16 GEMM -----------------------------------
// C[z][m][n] = sum_k A[z][m][k] * B[z][n][k]  (+ bias[n])
// A: row-major (lda), B: N-major/K-contiguous (ldb), C: row-major f32 (ldc).
// Block tile BM x BN, K-step 32, 8 waves, wave tile WM x WN.
// Full GEMMs: <256,128,64,64>  -> 16 WMMA per wave per K-step, 1 LDS ld/WMMA.
// Block-diag:  <128, 64,32,32> -> N=320 divisible tiling.
template <int BM, int BN, int WM, int WN>
__global__ __launch_bounds__(256) void k_gemm_bf16(
    const __bf16* __restrict__ A, int lda, long aZ,
    const __bf16* __restrict__ Bm, int ldb, long bZ,
    const float* __restrict__ bias,
    float* __restrict__ C, int ldc, long cZ,
    int K)
{
  constexpr int MF  = WM / 16;           // A fragments per wave
  constexpr int NF  = WN / 16;           // B fragments per wave
  constexpr int NWN = BN / WN;           // waves along N
  constexpr int ACH = BM / 64;           // A B128-chunks per thread per tile
  constexpr int BCH = BN / 64;           // B B128-chunks per thread per tile
  constexpr int INFLIGHT = ACH + BCH;    // async ops issued per tile per wave

  __shared__ __bf16 Ash[2 * BM * PAD];
  __shared__ __bf16 Bsh[2 * BN * PAD];

  const int  tid    = threadIdx.x;
  const int  z      = blockIdx.z;
  const long blockM = (long)blockIdx.y * BM;
  const int  blockN = blockIdx.x * BN;

  const __bf16* gA = A + (long)z * aZ + blockM * lda;
  const __bf16* gB = Bm + (long)z * bZ + (long)blockN * ldb;

  const unsigned ldsA0 = (unsigned)(size_t)(&Ash[0]);
  const unsigned ldsB0 = (unsigned)(size_t)(&Bsh[0]);

  const int lane = tid & 31;
  const int wid  = tid >> 5;
  const int wm0  = (wid / NWN) * WM;
  const int wn0  = (wid % NWN) * WN;
  const int hr   = lane & 15;
  const int ks   = (lane >> 4) * 8;      // K base select per lane half
  const int mo   = (lane >> 4) * 8;      // C-layout M offset per lane half

  auto issue = [&](int buf, int t) {
#pragma unroll
    for (int u = 0; u < ACH; ++u) {
      const int ch  = u * 256 + tid;     // 4 chunks per 32-elem row
      const int row = ch >> 2;
      const int col = (ch & 3) * 8;
      const __bf16* pa = gA + (size_t)row * lda + t * 32 + col;
      unsigned la = ldsA0 + (unsigned)((buf * BM + row) * PAD + col) * 2u;
      async_ld16(la, pa);
    }
#pragma unroll
    for (int u = 0; u < BCH; ++u) {
      const int ch  = u * 256 + tid;
      const int row = ch >> 2;
      const int col = (ch & 3) * 8;
      const __bf16* pb = gB + (size_t)row * ldb + t * 32 + col;
      unsigned lb = ldsB0 + (unsigned)((buf * BN + row) * PAD + col) * 2u;
      async_ld16(lb, pb);
    }
  };

  v8f acc[MF][NF];
#pragma unroll
  for (int i = 0; i < MF; ++i)
#pragma unroll
    for (int j = 0; j < NF; ++j) acc[i][j] = vzero();

  const int kT = K / 32;
  issue(0, 0);
  for (int t = 0; t < kT; ++t) {
    const int buf = t & 1;
    if (t + 1 < kT) { issue(buf ^ 1, t + 1); WAIT_ASYNC(INFLIGHT); }
    else            { WAIT_ASYNC(0); }
    __syncthreads();

    v16bf fa[MF], fb[NF];
#pragma unroll
    for (int i = 0; i < MF; ++i)
      fa[i] = ldfrag(&Ash[(buf * BM + wm0 + i * 16 + hr) * PAD + ks]);
#pragma unroll
    for (int j = 0; j < NF; ++j)
      fb[j] = ldfrag(&Bsh[(buf * BN + wn0 + j * 16 + hr) * PAD + ks]);

#pragma unroll
    for (int i = 0; i < MF; ++i)
#pragma unroll
      for (int j = 0; j < NF; ++j)
        acc[i][j] = __builtin_amdgcn_wmma_f32_16x16x32_bf16(
            false, fa[i], false, fb[j], (short)0, acc[i][j], false, false);

    __syncthreads();
  }

  // C/D layout: lanes 0-15: (M=r, N=lane); lanes 16-31: (M=8+r, N=lane-16)
  float* Cz = C + (long)z * cZ;
#pragma unroll
  for (int i = 0; i < MF; ++i) {
#pragma unroll
    for (int j = 0; j < NF; ++j) {
      const int  gn = blockN + wn0 + j * 16 + hr;
      const float bv = bias ? bias[gn] : 0.0f;
      const long gm = blockM + wm0 + i * 16 + mo;
#pragma unroll
      for (int r = 0; r < 8; ++r)
        Cz[(gm + r) * (long)ldc + gn] = acc[i][j][r] + bv;
    }
  }
}

// --------------------------- small helper kernels ----------------------------
__global__ void k_f32_to_bf16(const float* __restrict__ s, __bf16* __restrict__ d, long n) {
  long i  = (long)blockIdx.x * blockDim.x + threadIdx.x;
  long st = (long)gridDim.x * blockDim.x;
  for (; i < n; i += st) d[i] = (__bf16)s[i];
}

// (HEADS, i, j) f32 -> (HEADS, j, i) bf16 so the GEMM reads K-contiguous B.
__global__ void k_transpose_head(const float* __restrict__ w, __bf16* __restrict__ wt) {
  int i = blockIdx.x * blockDim.x + threadIdx.x;
  if (i >= HEADS_ * BW_ * BW_) return;
  int h = i / (BW_ * BW_);
  int r = i % (BW_ * BW_);
  int ii = r / BW_, jj = r % BW_;
  wt[(size_t)h * BW_ * BW_ + (size_t)jj * BW_ + ii] = (__bf16)w[i];
}

__global__ void k_conv(const float* __restrict__ xl, const float* __restrict__ cw,
                       const float* __restrict__ cb, float* __restrict__ xc,
                       __bf16* __restrict__ xcb) {
  const long n = (long)ML_ * LRU_;
  long i  = (long)blockIdx.x * blockDim.x + threadIdx.x;
  long st = (long)gridDim.x * blockDim.x;
  for (; i < n; i += st) {
    int  c  = (int)(i % LRU_);
    long bl = i / LRU_;
    int  l  = (int)(bl % L_);
    float acc = cb[c];
#pragma unroll
    for (int k = 0; k < DCONV_; ++k) {
      int ll = l - (DCONV_ - 1) + k;
      if (ll >= 0) acc += cw[c * DCONV_ + k] * xl[i + (long)(ll - l) * LRU_];
    }
    xc[i] = acc;
    xcb[i] = (__bf16)acc;
  }
}

__global__ void k_conv_state(const float* __restrict__ xl, const int* __restrict__ cidx,
                             float* __restrict__ out) {
  int i = blockIdx.x * blockDim.x + threadIdx.x;
  if (i >= B_ * LRU_ * DCONV_) return;
  int b = i / (LRU_ * DCONV_);
  int r = i % (LRU_ * DCONV_);
  int c = r / DCONV_;
  int idx = cidx[i];                 // index into zero-padded length L+3
  int l = idx - (DCONV_ - 1);
  out[i] = (l >= 0 && l < L_) ? xl[((size_t)b * L_ + l) * LRU_ + c] : 0.0f;
}

__global__ void k_gates(const float* gxI, const float* gaI, const float* __restrict__ xc,
                        const float* __restrict__ gxb, const float* __restrict__ gab,
                        const float* __restrict__ ap, const int* __restrict__ segpos,
                        float* aOut, float* nxOut) {
  const long n = (long)ML_ * LRU_;
  long i  = (long)blockIdx.x * blockDim.x + threadIdx.x;
  long st = (long)gridDim.x * blockDim.x;
  for (; i < n; i += st) {
    int  c  = (int)(i % LRU_);
    long bl = i / LRU_;
    float gateX = 1.0f / (1.0f + __expf(-(gxI[i] + gxb[c])));
    float gateA = 1.0f / (1.0f + __expf(-(gaI[i] + gab[c])));
    float apv = ap[c];
    float sp  = (apv > 20.0f) ? apv : log1pf(__expf(apv));
    float la  = -8.0f * sp * gateA;
    bool reset = (segpos[bl] == 0);
    float av   = reset ? 0.0f : __expf(la);
    float mult = reset ? 1.0f : sqrtf(fmaxf(0.0f, 1.0f - __expf(2.0f * la)));
    float nx   = xc[i] * gateX * mult;
    aOut[i]  = av;
    nxOut[i] = nx;
  }
}

// ----------------------- 3-pass chunked linear recurrence --------------------
__global__ void k_scan1(const float* __restrict__ a, const float* __restrict__ nx,
                        float* __restrict__ cA, float* __restrict__ cH) {
  int c  = blockIdx.y * 256 + threadIdx.x;
  int ch = blockIdx.x, b = blockIdx.z;
  size_t base = ((size_t)b * L_ + (size_t)ch * CLEN) * LRU_ + c;
  float pa = 1.0f, h = 0.0f;
  for (int i = 0; i < CLEN; ++i) {
    size_t idx = base + (size_t)i * LRU_;
    float av = a[idx];
    pa *= av;
    h = av * h + nx[idx];
  }
  size_t o = ((size_t)b * NCH + ch) * LRU_ + c;
  cA[o] = pa;
  cH[o] = h;
}

__global__ void k_scan2(const float* __restrict__ cA, const float* __restrict__ cH,
                        float* __restrict__ hinit) {
  int c = blockIdx.x * 256 + threadIdx.x;
  int b = blockIdx.y;
  float h = 0.0f;
  for (int ch = 0; ch < NCH; ++ch) {
    size_t o = ((size_t)b * NCH + ch) * LRU_ + c;
    hinit[o] = h;
    h = cA[o] * h + cH[o];
  }
}

__global__ void k_scan3(const float* __restrict__ a, const float* nx,
                        const float* __restrict__ hinit,
                        float* hout, float* __restrict__ lasth) {
  int c  = blockIdx.y * 256 + threadIdx.x;
  int ch = blockIdx.x, b = blockIdx.z;
  float h = hinit[((size_t)b * NCH + ch) * LRU_ + c];
  size_t base = ((size_t)b * L_ + (size_t)ch * CLEN) * LRU_ + c;
  for (int i = 0; i < CLEN; ++i) {
    size_t idx = base + (size_t)i * LRU_;
    h = a[idx] * h + nx[idx];
    hout[idx] = h;                 // in-place over nx is safe (read-then-write)
  }
  if (ch == NCH - 1) lasth[(size_t)b * LRU_ + c] = h;
}

__global__ void k_gelu_mul(const float* __restrict__ h, const float* __restrict__ y,
                           const float* __restrict__ yb, __bf16* __restrict__ t) {
  const long n = (long)ML_ * LRU_;
  long i  = (long)blockIdx.x * blockDim.x + threadIdx.x;
  long st = (long)gridDim.x * blockDim.x;
  for (; i < n; i += st) {
    int c = (int)(i % LRU_);
    float v = y[i] + yb[c];
    float g = 0.5f * v * (1.0f + erff(v * 0.70710678118654752f));
    t[i] = (__bf16)(h[i] * g);
  }
}

// --------------------------------- launch ------------------------------------
extern "C" void kernel_launch(void* const* d_in, const int* in_sizes, int n_in,
                              void* d_out, int out_size, void* d_ws, size_t ws_size,
                              hipStream_t stream) {
  (void)in_sizes; (void)n_in; (void)out_size; (void)ws_size;

  const float* x        = (const float*)d_in[0];
  const int*   segpos   = (const int*)d_in[1];
  const int*   conv_idx = (const int*)d_in[2];
  const float* Wx       = (const float*)d_in[3];
  const float* bx       = (const float*)d_in[4];
  const float* Wy       = (const float*)d_in[5];
  const float* conv_w   = (const float*)d_in[6];
  const float* conv_b   = (const float*)d_in[7];
  const float* gx_w     = (const float*)d_in[8];
  const float* gx_b     = (const float*)d_in[9];
  const float* ga_w     = (const float*)d_in[10];
  const float* ga_b     = (const float*)d_in[11];
  const float* a_param  = (const float*)d_in[12];
  const float* y_bias   = (const float*)d_in[13];
  const float* Wout     = (const float*)d_in[14];
  const float* bout     = (const float*)d_in[15];

  float* outp   = (float*)d_out;
  float* cstate = outp + (size_t)ML_ * W_;
  float* lasth  = cstate + (size_t)B_ * LRU_ * DCONV_;

  char* ws = (char*)d_ws;
  size_t off = 0;
  auto take = [&](size_t n) -> void* {
    void* r = ws + off;
    off += n;
    off = (off + 255) & ~(size_t)255;
    return r;
  };
  __bf16* xb    = (__bf16*)take((size_t)ML_ * W_ * 2);     // also reused for t (pre-Wout)
  __bf16* wxb   = (__bf16*)take((size_t)LRU_ * W_ * 2);
  __bf16* wyb   = (__bf16*)take((size_t)LRU_ * W_ * 2);
  __bf16* woutb = (__bf16*)take((size_t)W_ * LRU_ * 2);
  __bf16* gxwT  = (__bf16*)take((size_t)HEADS_ * BW_ * BW_ * 2);
  __bf16* gawT  = (__bf16*)take((size_t)HEADS_ * BW_ * BW_ * 2);
  float*  xlb   = (float*)take((size_t)ML_ * LRU_ * 4);
  float*  yb    = (float*)take((size_t)ML_ * LRU_ * 4);
  float*  xcb_f = (float*)take((size_t)ML_ * LRU_ * 4);
  __bf16* xcb   = (__bf16*)take((size_t)ML_ * LRU_ * 2);
  float*  gxb_f = (float*)take((size_t)ML_ * LRU_ * 4);    // then nx, then h (in place)
  float*  gab_f = (float*)take((size_t)ML_ * LRU_ * 4);    // then a (in place)
  float*  cA    = (float*)take((size_t)B_ * NCH * LRU_ * 4);
  float*  cH    = (float*)take((size_t)B_ * NCH * LRU_ * 4);
  float*  hinit = (float*)take((size_t)B_ * NCH * LRU_ * 4);

  // 1) precision conversion / weight transpose
  k_f32_to_bf16<<<4096, 256, 0, stream>>>(x,    xb,    (long)ML_ * W_);
  k_f32_to_bf16<<<2048, 256, 0, stream>>>(Wx,   wxb,   (long)LRU_ * W_);
  k_f32_to_bf16<<<2048, 256, 0, stream>>>(Wy,   wyb,   (long)LRU_ * W_);
  k_f32_to_bf16<<<2048, 256, 0, stream>>>(Wout, woutb, (long)W_ * LRU_);
  int nT = (HEADS_ * BW_ * BW_ + 255) / 256;
  k_transpose_head<<<nT, 256, 0, stream>>>(gx_w, gxwT);
  k_transpose_head<<<nT, 256, 0, stream>>>(ga_w, gawT);

  // 2) xl = x@Wx^T + bx ; y = x@Wy^T   (WMMA bf16, f32 accumulate)
  dim3 gFull(W_ / 128, ML_ / 256, 1);
  k_gemm_bf16<256, 128, 64, 64><<<gFull, 256, 0, stream>>>(
      xb, W_, 0, wxb, W_, 0, bx,      xlb, LRU_, 0, W_);
  k_gemm_bf16<256, 128, 64, 64><<<gFull, 256, 0, stream>>>(
      xb, W_, 0, wyb, W_, 0, nullptr, yb,  LRU_, 0, W_);

  // 3) depthwise causal conv + conv_state output
  k_conv<<<8192, 256, 0, stream>>>(xlb, conv_w, conv_b, xcb_f, xcb);
  k_conv_state<<<(B_ * LRU_ * DCONV_ + 255) / 256, 256, 0, stream>>>(xlb, conv_idx, cstate);

  // 4) block-diagonal gate GEMMs (per head via blockIdx.z strides)
  dim3 gBD(BW_ / 64, ML_ / 128, HEADS_);
  k_gemm_bf16<128, 64, 32, 32><<<gBD, 256, 0, stream>>>(
      xcb, LRU_, (long)BW_, gxwT, BW_, (long)BW_ * BW_,
      nullptr, gxb_f, LRU_, (long)BW_, BW_);
  k_gemm_bf16<128, 64, 32, 32><<<gBD, 256, 0, stream>>>(
      xcb, LRU_, (long)BW_, gawT, BW_, (long)BW_ * BW_,
      nullptr, gab_f, LRU_, (long)BW_, BW_);

  // 5) gates -> a (over ga buffer), nx (over gx buffer)
  k_gates<<<8192, 256, 0, stream>>>(gxb_f, gab_f, xcb_f, gx_b, ga_b, a_param, segpos,
                                    gab_f, gxb_f);

  // 6) chunked scan: h over the nx buffer, last_h to output
  k_scan1<<<dim3(NCH, LRU_ / 256, B_), 256, 0, stream>>>(gab_f, gxb_f, cA, cH);
  k_scan2<<<dim3(LRU_ / 256, B_), 256, 0, stream>>>(cA, cH, hinit);
  k_scan3<<<dim3(NCH, LRU_ / 256, B_), 256, 0, stream>>>(gab_f, gxb_f, hinit, gxb_f, lasth);

  // 7) t = h * gelu(y + y_bias) -> bf16 (reuse xb)
  k_gelu_mul<<<8192, 256, 0, stream>>>(gxb_f, yb, y_bias, xb);

  // 8) out = t@Wout^T + bout
  k_gemm_bf16<256, 128, 64, 64><<<dim3(W_ / 128, ML_ / 256, 1), 256, 0, stream>>>(
      xb, LRU_, 0, woutb, LRU_, 0, bout, outp, W_, 0, LRU_);
}
